// MultiHeadedAttentionNonRoped_48095043781141
// MI455X (gfx1250) — matrix operand order repaired and checked by reading, not compiled, not run on hardware
//
#include <hip/hip_runtime.h>

#define D_MODEL 1024
#define NUM_HEADS 16
#define BATCH 2
#define SEQ 2048
#define DH 64
#define BS (BATCH * SEQ) /* 4096 */

#define __AS1 __attribute__((address_space(1)))
#define __AS3 __attribute__((address_space(3)))

typedef __attribute__((ext_vector_type(16))) __bf16 v16bf;
typedef __attribute__((ext_vector_type(8)))  float  v8f;
typedef __attribute__((ext_vector_type(4)))  unsigned int u32x4;
typedef __attribute__((ext_vector_type(4)))  int i32x4;
typedef __attribute__((ext_vector_type(8)))  int i32x8;

#if __has_builtin(__builtin_amdgcn_tensor_load_to_lds) && __has_builtin(__builtin_amdgcn_s_wait_tensorcnt)
#define USE_TDM 1
#else
#define USE_TDM 0
#endif

union Frag {
  v16bf bf;
  u32x4 u4[2];
};

__device__ __forceinline__ unsigned short f2bf(float x) {
  unsigned int u = __float_as_uint(x);
  unsigned int r = (u + 0x7FFFu + ((u >> 16) & 1u)) >> 16;  // RNE
  return (unsigned short)r;
}

__device__ __forceinline__ v8f wmma_bf16(const Frag& a, const Frag& b, v8f c) {
  return __builtin_amdgcn_wmma_f32_16x16x32_bf16(
      false, a.bf, false, b.bf, (short)0, c, false, false);
}

// ---- async global->LDS copy of one 16-byte chunk (ASYNCcnt path), with fallback ----
__device__ __forceinline__ void async_cp16(unsigned short* lds_dst, const unsigned short* g_src) {
#if __has_builtin(__builtin_amdgcn_global_load_async_to_lds_b128)
  __builtin_amdgcn_global_load_async_to_lds_b128(
      (__AS1 i32x4*)g_src, (__AS3 i32x4*)lds_dst, 0, 0);
#else
  *(u32x4*)lds_dst = *(const u32x4*)g_src;
#endif
}

__device__ __forceinline__ void async_wait0() {
#if __has_builtin(__builtin_amdgcn_s_wait_asynccnt)
  __builtin_amdgcn_s_wait_asynccnt(0);
#endif
}

__device__ __forceinline__ void tensor_wait0() {
#if __has_builtin(__builtin_amdgcn_s_wait_tensorcnt)
  __builtin_amdgcn_s_wait_tensorcnt(0);
#endif
}

__device__ __forceinline__ unsigned lds_addr32(const void* p) {
  return (unsigned)(unsigned long long)(__AS3 const void*)p;
}

#if USE_TDM
// One 2D TDM descriptor load: tile (tile_d0 x tile_d1 rows) of a row-major
// bf16 tensor with row length tensor_d0 / row stride stride_d0 (elements),
// into LDS at lds_byte with per-row padding (pad codes per D# group1 spec).
__device__ __forceinline__ void tdm_load_2d(
    unsigned lds_byte, const unsigned short* gaddr,
    unsigned tensor_d0, unsigned tensor_d1, unsigned stride_d0,
    unsigned tile_d0, unsigned tile_d1,
    unsigned pad_interval_code, unsigned pad_amount_code) {
  unsigned long long ga = (unsigned long long)gaddr;
  u32x4 g0;
  g0.x = 1u;                                               // count=1, user D#
  g0.y = lds_byte;                                         // lds_addr
  g0.z = (unsigned)(ga & 0xffffffffu);                     // global_addr lo
  g0.w = (unsigned)((ga >> 32) & 0x01ffffffu) | (2u << 30);// global_addr hi | type=2
  i32x8 g1;
  g1[0] = (int)((1u << 16)                  // data_size = 2 bytes
                | (1u << 20)                // pad_enable
                | (pad_interval_code << 22)
                | (pad_amount_code << 25));
  g1[1] = (int)((tensor_d0 & 0xffffu) << 16);
  g1[2] = (int)(((tensor_d0 >> 16) & 0xffffu) | ((tensor_d1 & 0xffffu) << 16));
  g1[3] = (int)(((tensor_d1 >> 16) & 0xffffu) | ((tile_d0 & 0xffffu) << 16));
  g1[4] = (int)(tile_d1 & 0xffffu);         // tile_dim1; tile_dim2 = 0
  g1[5] = (int)stride_d0;                   // tensor_dim0_stride lo
  g1[6] = 0;
  g1[7] = 0;
  i32x4 z4 = {0, 0, 0, 0};
  i32x8 z8 = {0, 0, 0, 0, 0, 0, 0, 0};
  __builtin_amdgcn_tensor_load_to_lds(g0, g1, z4, z4, z8, 0);
}
#endif

// A-fragment (16x32, row M = lane&15; lanes 0-15: K {0..7,16..23}, lanes 16-31: K {8..15,24..31})
__device__ __forceinline__ void frag_a(Frag& f, const unsigned short* p) {
  f.u4[0] = *(const u32x4*)(p);
  f.u4[1] = *(const u32x4*)(p + 16);
}
// B-fragment (32x16 = M2^T; lane holds col N=lane&15; lanes 0-15: K {0..7,16..23}, lanes 16-31: K {8..15,24..31})
__device__ __forceinline__ void frag_b(Frag& f, const unsigned short* p) {
  f.u4[0] = *(const u32x4*)(p);
  f.u4[1] = *(const u32x4*)(p + 16);
}

__global__ void cvt_f32_bf16_kernel(const float* __restrict__ in,
                                    unsigned short* __restrict__ out, int n) {
  int i = blockIdx.x * blockDim.x + threadIdx.x;
  if (i < n) out[i] = f2bf(in[i]);
}

// ---------------- Q/K/V projection: Y = X * W^T ----------------
// grid (D/64, BS/128, 3); block 256 = 8 waves; wave tile 16(m) x 64(n); K-step 64.
// Tiles staged in LDS by TDM (wave 0, double-buffered), fallback async copies.
__global__ __launch_bounds__(256) void gemm_qkv_kernel(
    const unsigned short* __restrict__ Xb,
    const unsigned short* __restrict__ Wq, const unsigned short* __restrict__ Wk,
    const unsigned short* __restrict__ Wv,
    unsigned short* __restrict__ Qo, unsigned short* __restrict__ Ko,
    unsigned short* __restrict__ Vto) {
  __shared__ __align__(16) unsigned short As[2][128 * 72];
  __shared__ __align__(16) unsigned short Bsh[2][64 * 72];

  const int tid = threadIdx.x;
  const int lane = tid & 31, wave = tid >> 5;
  const int which = blockIdx.z;
  const unsigned short* W = (which == 0) ? Wq : ((which == 1) ? Wk : Wv);
  const int mblock = blockIdx.y * 128;
  const int nbase = blockIdx.x * 64;

  auto stage = [&](int buf, int k0) {
#if USE_TDM
    if (wave == 0) {
      // A tile 64(k) x 128 rows; row 128B = 32 DW -> pad code 4, +4 DW -> code 3
      tdm_load_2d(lds_addr32(&As[buf][0]), Xb + (size_t)mblock * D_MODEL + k0,
                  D_MODEL, BS, D_MODEL, 64, 128, 4, 3);
      // B tile 64(k) x 64 rows
      tdm_load_2d(lds_addr32(&Bsh[buf][0]), W + (size_t)nbase * D_MODEL + k0,
                  D_MODEL, D_MODEL, D_MODEL, 64, 64, 4, 3);
    }
#else
#pragma unroll
    for (int i = 0; i < 4; ++i) {
      int c = tid + i * 256;
      int row = c >> 3, col8 = (c & 7) * 8;
      async_cp16(&As[buf][row * 72 + col8],
                 Xb + (size_t)(mblock + row) * D_MODEL + k0 + col8);
    }
#pragma unroll
    for (int i = 0; i < 2; ++i) {
      int c = tid + i * 256;
      int row = c >> 3, col8 = (c & 7) * 8;
      async_cp16(&Bsh[buf][row * 72 + col8],
                 W + (size_t)(nbase + row) * D_MODEL + k0 + col8);
    }
#endif
  };

  v8f acc[4] = {};
  int buf = 0;
  stage(0, 0);
  for (int k0 = 0; k0 < D_MODEL; k0 += 64) {
#if USE_TDM
    if (wave == 0) tensor_wait0();
#else
    async_wait0();
#endif
    __syncthreads();
    if (k0 + 64 < D_MODEL) stage(buf ^ 1, k0 + 64);
#pragma unroll
    for (int kc = 0; kc < 2; ++kc) {
      Frag a;
      frag_a(a, &As[buf][(wave * 16 + (lane & 15)) * 72 + kc * 32 + ((lane >> 4) << 3)]);
      Frag bfr[4];
#pragma unroll
      for (int j = 0; j < 4; ++j)
        frag_b(bfr[j], &Bsh[buf][(j * 16 + (lane & 15)) * 72 + kc * 32 + ((lane >> 4) << 3)]);
#pragma unroll
      for (int j = 0; j < 4; ++j)
        acc[j] = wmma_bf16(a, bfr[j], acc[j]);
    }
    buf ^= 1;
  }

  const int half = lane >> 4, nlo = lane & 15;
  const int mbase = mblock + wave * 16;
#pragma unroll
  for (int j = 0; j < 4; ++j) {
#pragma unroll
    for (int r = 0; r < 8; ++r) {
      int m = mbase + r + 8 * half;
      int n = nbase + j * 16 + nlo;
      int b = m >> 11, t = m & (SEQ - 1);
      int h = n >> 6, d = n & (DH - 1);
      unsigned short v = f2bf(acc[j][r]);
      if (which == 0)      Qo[(((size_t)(b * NUM_HEADS + h) * SEQ) + t) * DH + d] = v;
      else if (which == 1) Ko[(((size_t)(b * NUM_HEADS + h) * SEQ) + t) * DH + d] = v;
      else                 Vto[(((size_t)(b * NUM_HEADS + h) * DH) + d) * SEQ + t] = v;
    }
  }
}

// ---------------- causal flash attention ----------------
// grid (SEQ/64, B*H); block 128 = 4 waves; wave = 16 query rows; key tile 32.
// K/V tiles staged in LDS by TDM (wave 0, double-buffered), shared by 4 waves.
__global__ __launch_bounds__(128) void attn_kernel(
    const unsigned short* __restrict__ Q, const unsigned short* __restrict__ K,
    const unsigned short* __restrict__ Vt, unsigned short* __restrict__ Ctx) {
  __shared__ __align__(16) unsigned short Ks[2][32 * 72];
  __shared__ __align__(16) unsigned short Vs[2][64 * 40];
  __shared__ __align__(16) unsigned short plds[4][16 * 32];

  const int tid = threadIdx.x;
  const int lane = tid & 31, wave = tid >> 5;
  const int bh = blockIdx.y;
  const int b = bh >> 4, h = bh & 15;
  const int qbase = blockIdx.x * 64 + wave * 16;
  const int half = lane >> 4, nlo = lane & 15;

  const unsigned short* Qh = Q  + (size_t)(b * NUM_HEADS + h) * SEQ * DH;
  const unsigned short* Kh = K  + (size_t)(b * NUM_HEADS + h) * SEQ * DH;
  const unsigned short* Vh = Vt + (size_t)(b * NUM_HEADS + h) * DH * SEQ;

  auto stage = [&](int buf, int kbase) {
#if USE_TDM
    if (wave == 0) {
      // K tile: 64(d) x 32 key rows, row 128B = 32 DW -> pad codes (4, 3)
      tdm_load_2d(lds_addr32(&Ks[buf][0]), Kh + (size_t)kbase * DH,
                  DH, SEQ, DH, DH, 32, 4, 3);
      // Vt tile: 32(t) x 64 d-rows, row 64B = 16 DW -> pad codes (3, 3)
      tdm_load_2d(lds_addr32(&Vs[buf][0]), Vh + kbase,
                  SEQ, DH, SEQ, 32, DH, 3, 3);
    }
#else
#pragma unroll
    for (int i = 0; i < 2; ++i) {
      int c = tid + i * 128;
      int row = c >> 3, col8 = (c & 7) * 8;
      async_cp16(&Ks[buf][row * 72 + col8],
                 Kh + (size_t)(kbase + row) * DH + col8);
    }
#pragma unroll
    for (int i = 0; i < 2; ++i) {
      int c = tid + i * 128;
      int row = c >> 2, col8 = (c & 3) * 8;
      async_cp16(&Vs[buf][row * 40 + col8],
                 Vh + (size_t)row * SEQ + kbase + col8);
    }
#endif
  };

  Frag aq[2];
  frag_a(aq[0], Qh + (size_t)(qbase + (lane & 15)) * DH + 0  + ((lane >> 4) << 3));
  frag_a(aq[1], Qh + (size_t)(qbase + (lane & 15)) * DH + 32 + ((lane >> 4) << 3));

  v8f acc[4] = {};
  float mrow[8], lrow[8];
#pragma unroll
  for (int r = 0; r < 8; ++r) { mrow[r] = -3.0e38f; lrow[r] = 0.0f; }

  const int ktiles = blockIdx.x * 2 + 2;  // uniform across block; masked tiles are no-ops
  int buf = 0;
  stage(0, 0);
  for (int kt = 0; kt < ktiles; ++kt) {
    const int kbase = kt * 32;
#if USE_TDM
    if (wave == 0) tensor_wait0();
#else
    async_wait0();
#endif
    __syncthreads();
    if (kt + 1 < ktiles) stage(buf ^ 1, kbase + 32);

    // S(16x32) = Q(16x64) * K_tile^T : load all 4 B-frags, then 4 WMMAs
    Frag bk[4];
#pragma unroll
    for (int kc = 0; kc < 2; ++kc)
#pragma unroll
      for (int j = 0; j < 2; ++j)
        frag_b(bk[kc * 2 + j],
               &Ks[buf][(j * 16 + (lane & 15)) * 72 + kc * 32 + ((lane >> 4) << 3)]);
    v8f sc[2] = {};
#pragma unroll
    for (int kc = 0; kc < 2; ++kc)
#pragma unroll
      for (int j = 0; j < 2; ++j)
        sc[j] = wmma_bf16(aq[kc], bk[kc * 2 + j], sc[j]);

    // mask + scale + online softmax
    float lg[2][8], newm[8];
#pragma unroll
    for (int r = 0; r < 8; ++r) newm[r] = mrow[r];
#pragma unroll
    for (int j = 0; j < 2; ++j) {
      int kcol = kbase + j * 16 + nlo;
#pragma unroll
      for (int r = 0; r < 8; ++r) {
        int qrow = qbase + r + 8 * half;
        float v = sc[j][r] * 0.125f;
        if (kcol > qrow) v = -3.0e38f;
        lg[j][r] = v;
        newm[r] = fmaxf(newm[r], v);
      }
    }
#pragma unroll
    for (int r = 0; r < 8; ++r) {
      float v = newm[r];
      v = fmaxf(v, __shfl_xor(v, 1, 16));
      v = fmaxf(v, __shfl_xor(v, 2, 16));
      v = fmaxf(v, __shfl_xor(v, 4, 16));
      v = fmaxf(v, __shfl_xor(v, 8, 16));
      newm[r] = v;
    }

    float alpha[8], psum[8];
    unsigned short pb[2][8];
#pragma unroll
    for (int r = 0; r < 8; ++r) {
      alpha[r] = __expf(mrow[r] - newm[r]);
      psum[r] = 0.0f;
    }
#pragma unroll
    for (int j = 0; j < 2; ++j) {
#pragma unroll
      for (int r = 0; r < 8; ++r) {
        float p = __expf(lg[j][r] - newm[r]);
        psum[r] += p;
        pb[j][r] = f2bf(p);
      }
    }
#pragma unroll
    for (int r = 0; r < 8; ++r) {
      float v = psum[r];
      v += __shfl_xor(v, 1, 16);
      v += __shfl_xor(v, 2, 16);
      v += __shfl_xor(v, 4, 16);
      v += __shfl_xor(v, 8, 16);
      lrow[r] = lrow[r] * alpha[r] + v;
      mrow[r] = newm[r];
    }
#pragma unroll
    for (int j = 0; j < 4; ++j)
#pragma unroll
      for (int r = 0; r < 8; ++r)
        acc[j][r] *= alpha[r];

    // re-layout P (C-layout) -> A-fragment via wave-private LDS
#pragma unroll
    for (int j = 0; j < 2; ++j)
#pragma unroll
      for (int r = 0; r < 8; ++r)
        plds[wave][(r + 8 * half) * 32 + j * 16 + nlo] = pb[j][r];
    __syncthreads();
    Frag ap;
    frag_a(ap, &plds[wave][(lane & 15) * 32 + ((lane >> 4) << 3)]);

    // ctx(16x64) += P(16x32) * V_tile(32x64) : load all 4 B-frags, then 4 WMMAs
    Frag bv[4];
#pragma unroll
    for (int j = 0; j < 4; ++j)
      frag_b(bv[j], &Vs[buf][(j * 16 + (lane & 15)) * 40 + ((lane >> 4) << 3)]);
#pragma unroll
    for (int j = 0; j < 4; ++j)
      acc[j] = wmma_bf16(ap, bv[j], acc[j]);
    buf ^= 1;
  }

  // finalize: ctx /= l, store concat layout (b, t, h*64 + d) as bf16
#pragma unroll
  for (int j = 0; j < 4; ++j) {
#pragma unroll
    for (int r = 0; r < 8; ++r) {
      int qrow = qbase + r + 8 * half;
      int d = j * 16 + nlo;
      float v = acc[j][r] / lrow[r];
      Ctx[((size_t)(b * SEQ + qrow)) * D_MODEL + h * DH + d] = f2bf(v);
    }
  }
}

// ---------------- output projection: Out = Ctx * Wo^T (fp32 out) ----------------
__global__ __launch_bounds__(256) void gemm_o_kernel(
    const unsigned short* __restrict__ Cb, const unsigned short* __restrict__ Wo,
    float* __restrict__ Out) {
  __shared__ __align__(16) unsigned short As[2][128 * 72];
  __shared__ __align__(16) unsigned short Bsh[2][64 * 72];

  const int tid = threadIdx.x;
  const int lane = tid & 31, wave = tid >> 5;
  const int mblock = blockIdx.y * 128;
  const int nbase = blockIdx.x * 64;

  auto stage = [&](int buf, int k0) {
#if USE_TDM
    if (wave == 0) {
      tdm_load_2d(lds_addr32(&As[buf][0]), Cb + (size_t)mblock * D_MODEL + k0,
                  D_MODEL, BS, D_MODEL, 64, 128, 4, 3);
      tdm_load_2d(lds_addr32(&Bsh[buf][0]), Wo + (size_t)nbase * D_MODEL + k0,
                  D_MODEL, D_MODEL, D_MODEL, 64, 64, 4, 3);
    }
#else
#pragma unroll
    for (int i = 0; i < 4; ++i) {
      int c = tid + i * 256;
      int row = c >> 3, col8 = (c & 7) * 8;
      async_cp16(&As[buf][row * 72 + col8],
                 Cb + (size_t)(mblock + row) * D_MODEL + k0 + col8);
    }
#pragma unroll
    for (int i = 0; i < 2; ++i) {
      int c = tid + i * 256;
      int row = c >> 3, col8 = (c & 7) * 8;
      async_cp16(&Bsh[buf][row * 72 + col8],
                 Wo + (size_t)(nbase + row) * D_MODEL + k0 + col8);
    }
#endif
  };

  v8f acc[4] = {};
  int buf = 0;
  stage(0, 0);
  for (int k0 = 0; k0 < D_MODEL; k0 += 64) {
#if USE_TDM
    if (wave == 0) tensor_wait0();
#else
    async_wait0();
#endif
    __syncthreads();
    if (k0 + 64 < D_MODEL) stage(buf ^ 1, k0 + 64);
#pragma unroll
    for (int kc = 0; kc < 2; ++kc) {
      Frag a;
      frag_a(a, &As[buf][(wave * 16 + (lane & 15)) * 72 + kc * 32 + ((lane >> 4) << 3)]);
      Frag bfr[4];
#pragma unroll
      for (int j = 0; j < 4; ++j)
        frag_b(bfr[j], &Bsh[buf][(j * 16 + (lane & 15)) * 72 + kc * 32 + ((lane >> 4) << 3)]);
#pragma unroll
      for (int j = 0; j < 4; ++j)
        acc[j] = wmma_bf16(a, bfr[j], acc[j]);
    }
    buf ^= 1;
  }

  const int half = lane >> 4, nlo = lane & 15;
  const int mbase = mblock + wave * 16;
#pragma unroll
  for (int j = 0; j < 4; ++j) {
#pragma unroll
    for (int r = 0; r < 8; ++r) {
      int m = mbase + r + 8 * half;
      int n = nbase + j * 16 + nlo;
      Out[(size_t)m * D_MODEL + n] = acc[j][r];
    }
  }
}

extern "C" void kernel_launch(void* const* d_in, const int* in_sizes, int n_in,
                              void* d_out, int out_size, void* d_ws, size_t ws_size,
                              hipStream_t stream) {
  const float* wq = (const float*)d_in[0];
  const float* wk = (const float*)d_in[1];
  const float* wv = (const float*)d_in[2];
  const float* wo = (const float*)d_in[3];
  const float* x  = (const float*)d_in[4];

  unsigned short* ws0 = (unsigned short*)d_ws;
  size_t o = 0;
  unsigned short* Xb  = ws0 + o; o += (size_t)BS * D_MODEL;
  unsigned short* Wqb = ws0 + o; o += (size_t)D_MODEL * D_MODEL;
  unsigned short* Wkb = ws0 + o; o += (size_t)D_MODEL * D_MODEL;
  unsigned short* Wvb = ws0 + o; o += (size_t)D_MODEL * D_MODEL;
  unsigned short* Wob = ws0 + o; o += (size_t)D_MODEL * D_MODEL;
  unsigned short* Qb  = ws0 + o; o += (size_t)BATCH * NUM_HEADS * SEQ * DH;
  unsigned short* Kb  = ws0 + o; o += (size_t)BATCH * NUM_HEADS * SEQ * DH;
  unsigned short* Vtb = ws0 + o; o += (size_t)BATCH * NUM_HEADS * SEQ * DH;
  unsigned short* Cb  = Xb;  // Xb dead after QKV projections -> reuse for concat ctx

  const int nX = BS * D_MODEL;
  const int nW = D_MODEL * D_MODEL;
  cvt_f32_bf16_kernel<<<(nX + 255) / 256, 256, 0, stream>>>(x, Xb, nX);
  cvt_f32_bf16_kernel<<<(nW + 255) / 256, 256, 0, stream>>>(wq, Wqb, nW);
  cvt_f32_bf16_kernel<<<(nW + 255) / 256, 256, 0, stream>>>(wk, Wkb, nW);
  cvt_f32_bf16_kernel<<<(nW + 255) / 256, 256, 0, stream>>>(wv, Wvb, nW);
  cvt_f32_bf16_kernel<<<(nW + 255) / 256, 256, 0, stream>>>(wo, Wob, nW);

  gemm_qkv_kernel<<<dim3(D_MODEL / 64, BS / 128, 3), 256, 0, stream>>>(
      Xb, Wqb, Wkb, Wvb, Qb, Kb, Vtb);

  attn_kernel<<<dim3(SEQ / 64, BATCH * NUM_HEADS), 128, 0, stream>>>(Qb, Kb, Vtb, Cb);

  gemm_o_kernel<<<dim3(D_MODEL / 64, BS / 128), 256, 0, stream>>>(Cb, Wob, (float*)d_out);
}
